// PoolingReformer_31645319037446
// MI455X (gfx1250) — matrix operand order repaired and checked by reading, not compiled, not run on hardware
//
#include <hip/hip_runtime.h>
#include <math.h>

#define Bn   2
#define Tn   4096
#define Dn   1024
#define Hn   8
#define DHn  128
#define NHn  8
#define NBn  64
#define NCn  (NHn * NBn)   /* 512 chunks */
#define BHn  (Bn * Hn)     /* 16 */
#define FFn  4096
#define Rn   (NBn / 2)     /* 32 rotations */
#define Ln   2

typedef __attribute__((ext_vector_type(16))) __bf16          v16bf;
typedef __attribute__((ext_vector_type(16))) unsigned short  v16us;
typedef __attribute__((ext_vector_type(8)))  float           v8f;
typedef __attribute__((ext_vector_type(8)))  unsigned short  us8;   // 16-byte vector
typedef __attribute__((ext_vector_type(4)))  unsigned short  us4;   // 8-byte vector
typedef __attribute__((ext_vector_type(4)))  float           f4;    // 16-byte vector
typedef __attribute__((ext_vector_type(4)))  int             i4;    // 16-byte vector

__device__ __forceinline__ unsigned short f2bf(float f) {
  unsigned u = __float_as_uint(f);
  unsigned r = u + 0x7FFFu + ((u >> 16) & 1u);
  return (unsigned short)(r >> 16);
}
__device__ __forceinline__ float bf2f(unsigned short h) {
  return __uint_as_float(((unsigned)h) << 16);
}

__device__ __forceinline__ v8f wmma_bf16(v16bf a, v16bf b, v8f c) {
  return __builtin_amdgcn_wmma_f32_16x16x32_bf16(false, a, false, b, (short)0, c, false, false);
}

// A fragment: src is M x K row-major bf16 (as ushort) in LDS.
// A 16x32 layout (contiguous, matches B fragments): lanes 0-15 K 0..15, lanes 16-31 K 16..31
__device__ __forceinline__ v16bf fragA(const unsigned short* S, int ld, int mBase, int kBase, int lane) {
  int m  = mBase + (lane & 15);
  int kb = kBase + ((lane & 16) ? 16 : 0);
  v16us t;
#pragma unroll
  for (int v = 0; v < 8; ++v) {
    int k = kb + 2 * v;
    t[2 * v]     = S[m * ld + k];
    t[2 * v + 1] = S[m * ld + k + 1];
  }
  return __builtin_bit_cast(v16bf, t);
}
// B fragment from K x N row-major storage. Lanes 0-15: K 0..15, lanes 16-31: K 16..31.
__device__ __forceinline__ v16bf fragB_KN(const unsigned short* S, int ld, int kBase, int nBase, int lane) {
  int n  = nBase + (lane & 15);
  int kb = kBase + ((lane & 16) ? 16 : 0);
  v16us t;
#pragma unroll
  for (int v = 0; v < 8; ++v) {
    int k = kb + 2 * v;
    t[2 * v]     = S[k * ld + n];
    t[2 * v + 1] = S[(k + 1) * ld + n];
  }
  return __builtin_bit_cast(v16bf, t);
}
// B fragment where source rows are the N dimension (N x K row-major, e.g. key rows)
__device__ __forceinline__ v16bf fragB_NK(const unsigned short* S, int ld, int nBase, int kBase, int lane) {
  int n  = nBase + (lane & 15);
  int kb = kBase + ((lane & 16) ? 16 : 0);
  v16us t;
#pragma unroll
  for (int v = 0; v < 8; ++v) {
    t[2 * v]     = S[n * ld + kb + 2 * v];
    t[2 * v + 1] = S[n * ld + kb + 2 * v + 1];
  }
  return __builtin_bit_cast(v16bf, t);
}

// ---------------------------------------------------------------------------
// GEMM: C(MxN) = A(MxK) @ B(KxN), A/B bf16, f32 accumulate.
// EPI 1: +bias +residual -> f32 out ; EPI 2: plain -> bf16 out ; EPI 3: +bias, GELU -> bf16 out
// ---------------------------------------------------------------------------
template <int EPI>
__global__ __launch_bounds__(256) void k_gemm(const unsigned short* __restrict__ A,
                                              const unsigned short* __restrict__ Bw,
                                              const float* __restrict__ bias,
                                              const float* __restrict__ resid,
                                              float* __restrict__ Cf,
                                              unsigned short* __restrict__ Cb,
                                              int M, int N, int K) {
  __shared__ __align__(16) unsigned short As[128 * 40];
  __shared__ __align__(16) unsigned short Bs[32 * 136];
  const int tid = threadIdx.x, lane = tid & 31, wid = tid >> 5;
  const int wm = wid & 3, wn = wid >> 2;
  const int tM = blockIdx.y * 128, tN = blockIdx.x * 128;
  v8f acc[2][4];
  for (int mi = 0; mi < 2; ++mi)
    for (int ni = 0; ni < 4; ++ni)
      for (int e = 0; e < 8; ++e) acc[mi][ni][e] = 0.f;

  for (int k0 = 0; k0 < K; k0 += 32) {
    // 16-byte vector staging: 128x32 A tile, 32x128 B tile
    for (int i = tid * 8; i < 128 * 32; i += 256 * 8) {
      int r = i >> 5, c = i & 31;
      *(us8*)&As[r * 40 + c] = *(const us8*)&A[(size_t)(tM + r) * K + k0 + c];
    }
    for (int i = tid * 8; i < 32 * 128; i += 256 * 8) {
      int r = i >> 7, c = i & 127;
      *(us8*)&Bs[r * 136 + c] = *(const us8*)&Bw[(size_t)(k0 + r) * N + tN + c];
    }
    __syncthreads();
    if (k0 + 32 < K) {  // prefetch next K tile (global_prefetch_b8)
      int ia = tid * 8;
      __builtin_prefetch(&A[(size_t)(tM + (ia >> 5)) * K + (k0 + 32) + (ia & 31)], 0, 1);
      __builtin_prefetch(&Bw[(size_t)(k0 + 32 + (ia >> 7)) * N + tN + (ia & 127)], 0, 1);
    }
    v16bf a0 = fragA(As, 40, wm * 32, 0, lane);
    v16bf a1 = fragA(As, 40, wm * 32 + 16, 0, lane);
#pragma unroll
    for (int ni = 0; ni < 4; ++ni) {
      v16bf bfr = fragB_KN(Bs, 136, 0, wn * 64 + ni * 16, lane);
      acc[0][ni] = wmma_bf16(a0, bfr, acc[0][ni]);
      acc[1][ni] = wmma_bf16(a1, bfr, acc[1][ni]);
    }
    __syncthreads();
  }

  const int half8 = (lane & 16) ? 8 : 0;
#pragma unroll
  for (int mi = 0; mi < 2; ++mi)
#pragma unroll
    for (int ni = 0; ni < 4; ++ni)
#pragma unroll
      for (int r = 0; r < 8; ++r) {
        int row = tM + wm * 32 + mi * 16 + r + half8;
        int col = tN + wn * 64 + ni * 16 + (lane & 15);
        float v = acc[mi][ni][r];
        size_t idx = (size_t)row * N + col;
        if (EPI == 1) {
          v += bias[col] + resid[idx];
          Cf[idx] = v;
        } else if (EPI == 2) {
          Cb[idx] = f2bf(v);
        } else {
          v += bias[col];
          v = 0.5f * v * (1.0f + erff(v * 0.70710678118654752f));  // exact GELU
          Cb[idx] = f2bf(v);
        }
      }
}

// ---------------------------------------------------------------------------
// LayerNorm: f32 in -> bf16 out, one block per row (D=1024, exactly one f4/thread)
// ---------------------------------------------------------------------------
__global__ __launch_bounds__(256) void k_ln(const float* __restrict__ X, const float* __restrict__ g,
                                            const float* __restrict__ bta, unsigned short* __restrict__ Y) {
  const int row = blockIdx.x;
  const float* x = X + (size_t)row * Dn;
  __shared__ float red[256];
  const int tid = threadIdx.x;
  f4 xv = *(const f4*)&x[tid * 4];
  red[tid] = xv[0] + xv[1] + xv[2] + xv[3];
  __syncthreads();
  for (int o = 128; o > 0; o >>= 1) { if (tid < o) red[tid] += red[tid + o]; __syncthreads(); }
  float mean = red[0] * (1.f / Dn);
  __syncthreads();
  float vs = 0.f;
#pragma unroll
  for (int e = 0; e < 4; ++e) { float d = xv[e] - mean; vs += d * d; }
  red[tid] = vs;
  __syncthreads();
  for (int o = 128; o > 0; o >>= 1) { if (tid < o) red[tid] += red[tid + o]; __syncthreads(); }
  float inv = rsqrtf(red[0] * (1.f / Dn) + 1e-5f);
  f4 gv = *(const f4*)&g[tid * 4];
  f4 bv = *(const f4*)&bta[tid * 4];
  us4 yv;
#pragma unroll
  for (int e = 0; e < 4; ++e) yv[e] = f2bf((xv[e] - mean) * inv * gv[e] + bv[e]);
  *(us4*)&Y[(size_t)row * Dn + tid * 4] = yv;
}

// ---------------------------------------------------------------------------
// LSH rotation + argmax over [v, -v]  -> bucket in [0,64)
// grid (T/64, NH, BH), block 64
// ---------------------------------------------------------------------------
__global__ __launch_bounds__(64) void k_buckets(const unsigned short* __restrict__ qk,
                                                const float* __restrict__ rot,
                                                int* __restrict__ buckets) {
  __shared__ float rs[DHn * Rn];
  const int h = blockIdx.y, bh = blockIdx.z;
  const int t = blockIdx.x * 64 + threadIdx.x;
  for (int i = threadIdx.x; i < DHn * Rn; i += 64) {
    int d = i / Rn, r = i % Rn;
    rs[i] = rot[((size_t)d * NHn + h) * Rn + r];
  }
  __syncthreads();
  const int b_ = bh / Hn, hd = bh % Hn;
  const us8* q8 = (const us8*)(qk + ((size_t)(b_ * Tn + t)) * (Hn * DHn) + hd * DHn);
  float acc[Rn];
#pragma unroll
  for (int r = 0; r < Rn; ++r) acc[r] = 0.f;
  for (int d8 = 0; d8 < DHn / 8; ++d8) {
    us8 vq = q8[d8];
#pragma unroll
    for (int e = 0; e < 8; ++e) {
      float qv = bf2f(vq[e]);
      const float* rr = &rs[(d8 * 8 + e) * Rn];
#pragma unroll
      for (int r = 0; r < Rn; ++r) acc[r] += qv * rr[r];
    }
  }
  float best = acc[0];
  int bi = 0;
#pragma unroll
  for (int r = 1; r < Rn; ++r) if (acc[r] > best) { best = acc[r]; bi = r; }
#pragma unroll
  for (int r = 0; r < Rn; ++r) if (-acc[r] > best) { best = -acc[r]; bi = Rn + r; }
  buckets[((size_t)bh * NHn + h) * Tn + t] = bi;
}

// ---------------------------------------------------------------------------
// Stable counting sort per (bh, hash): 4096 items, 64 bins, one 256-thread block.
// Keys (bucket, pos) are unique; hash regions disjoint -> exact reference argsort.
// ---------------------------------------------------------------------------
__global__ __launch_bounds__(256) void k_sort(const int* __restrict__ buckets,
                                              int* __restrict__ st, int* __restrict__ undo) {
  __shared__ unsigned short cnt[64 * 256];  // [bin][thread], 32KB
  __shared__ unsigned binStart[64];
  const int h = blockIdx.x, bh = blockIdx.y;
  const int* bk = buckets + ((size_t)bh * NHn + h) * Tn;
  const int tid = threadIdx.x;
  for (int b = 0; b < 64; ++b) cnt[b * 256 + tid] = 0;
  int loc[16];
  const i4* bk4 = (const i4*)(bk + tid * 16);
#pragma unroll
  for (int q = 0; q < 4; ++q) {
    i4 vb = bk4[q];
#pragma unroll
    for (int e = 0; e < 4; ++e) loc[q * 4 + e] = vb[e];
  }
  for (int k = 0; k < 16; ++k) cnt[loc[k] * 256 + tid]++;
  __syncthreads();
  if (tid < 64) {  // per-bin exclusive scan over thread columns
    unsigned run = 0;
    for (int i = 0; i < 256; ++i) {
      unsigned c = cnt[tid * 256 + i];
      cnt[tid * 256 + i] = (unsigned short)run;
      run += c;
    }
    binStart[tid] = run;  // temporarily the bin total
  }
  __syncthreads();
  if (tid == 0) {  // exclusive scan over 64 bins
    unsigned r = 0;
    for (int b = 0; b < 64; ++b) { unsigned tot = binStart[b]; binStart[b] = r; r += tot; }
  }
  __syncthreads();
  int* stR = st + (size_t)bh * (NHn * Tn);
  int* unR = undo + (size_t)bh * (NHn * Tn);
  for (int k = 0; k < 16; ++k) {
    int b = loc[k];
    int prior = 0;
    for (int j = 0; j < k; ++j) prior += (loc[j] == b) ? 1 : 0;
    int rank = (int)binStart[b] + (int)cnt[b * 256 + tid] + prior;
    int pos = tid * 16 + k;
    stR[h * Tn + rank] = pos;             // sorted slot -> original position
    unR[h * Tn + pos] = h * Tn + rank;    // (hash, pos) -> sorted slot
  }
}

// ---------------------------------------------------------------------------
// Attention pass 1: per chunk, dots = Q Khat^T (WMMA), mask, row logsumexp.
// grid (512, 16), block 128 (4 waves)
// ---------------------------------------------------------------------------
__global__ __launch_bounds__(128) void k_attn_lse(const unsigned short* __restrict__ qk,
                                                  const int* __restrict__ st,
                                                  float* __restrict__ lse) {
  __shared__ __align__(16) unsigned char smA[64 * 136 * 2];   // Q tile
  __shared__ __align__(16) unsigned char smB[128 * 136 * 2];  // K tile, then f32 scores
  __shared__ int qpos[64];
  __shared__ int kpos[128];
  unsigned short* Qs = (unsigned short*)smA;
  unsigned short* Ks = (unsigned short*)smB;
  const int c = blockIdx.x, bh = blockIdx.y;
  const int pc = (c + NCn - 1) & (NCn - 1);
  const int tid = threadIdx.x, lane = tid & 31, w = tid >> 5;
  const int b_ = bh >> 3, hd = bh & 7;
  const int* stR = st + (size_t)bh * (NHn * Tn);
  if (tid < 64) qpos[tid] = stR[c * 64 + tid];
  kpos[tid] = (tid < 64) ? stR[c * 64 + tid] : stR[pc * 64 + (tid - 64)];
  __syncthreads();
  if (tid < 64) {  // Q rows, 16B vectors
    const us8* src = (const us8*)(qk + ((size_t)(b_ * Tn + qpos[tid])) * (Hn * DHn) + hd * DHn);
    us8* dst = (us8*)&Qs[tid * 136];
#pragma unroll
    for (int d8 = 0; d8 < DHn / 8; ++d8) dst[d8] = src[d8];
  }
  {  // each thread loads + L2-normalizes one key row
    const us8* src = (const us8*)(qk + ((size_t)(b_ * Tn + kpos[tid])) * (Hn * DHn) + hd * DHn);
    us8* dst = (us8*)&Ks[tid * 136];
    float nrm = 0.f;
#pragma unroll
    for (int d8 = 0; d8 < DHn / 8; ++d8) {
      us8 vld = src[d8];
      dst[d8] = vld;
#pragma unroll
      for (int e = 0; e < 8; ++e) { float f = bf2f(vld[e]); nrm += f * f; }
    }
    float inv = 1.f / fmaxf(sqrtf(nrm), 1e-12f);
#pragma unroll
    for (int d8 = 0; d8 < DHn / 8; ++d8) {
      us8 vld = dst[d8];
#pragma unroll
      for (int e = 0; e < 8; ++e) vld[e] = f2bf(bf2f(vld[e]) * inv);
      dst[d8] = vld;
    }
  }
  __syncthreads();
  v8f acc[8];
  for (int nt = 0; nt < 8; ++nt)
    for (int e = 0; e < 8; ++e) acc[nt][e] = 0.f;
  for (int k0 = 0; k0 < DHn; k0 += 32) {
    v16bf a = fragA(Qs, 136, w * 16, k0, lane);
#pragma unroll
    for (int nt = 0; nt < 8; ++nt) {
      v16bf bfr = fragB_NK(Ks, 136, nt * 16, k0, lane);
      acc[nt] = wmma_bf16(a, bfr, acc[nt]);
    }
  }
  __syncthreads();  // everyone done reading Ks -> reuse as f32 score buffer
  float* Ss = (float*)smB;  // 64 x 132
  const int half8 = (lane & 16) ? 8 : 0;
  const float scale = 0.08838834764831845f;  // 128^-0.5
#pragma unroll
  for (int nt = 0; nt < 8; ++nt)
#pragma unroll
    for (int r = 0; r < 8; ++r) {
      int i = w * 16 + r + half8;
      int j = nt * 16 + (lane & 15);
      float d = acc[nt][r] * scale;
      if (qpos[i] < kpos[j]) d = -1e9f;
      if (qpos[i] == kpos[j]) d = -5e4f;
      Ss[i * 132 + j] = d;
    }
  __syncthreads();
  if (tid < 64) {
    float m = -3.0e38f;
    for (int j = 0; j < 128; ++j) m = fmaxf(m, Ss[tid * 132 + j]);
    float z = 0.f;
    for (int j = 0; j < 128; ++j) z += expf(Ss[tid * 132 + j] - m);
    lse[(size_t)bh * (NHn * Tn) + c * 64 + tid] = m + logf(z);
  }
}

// ---------------------------------------------------------------------------
// Per-position softmax over the NH hash rounds -> probability per sorted slot
// grid (T/256, BH), block 256
// ---------------------------------------------------------------------------
__global__ __launch_bounds__(256) void k_combine(const float* __restrict__ lse,
                                                 const int* __restrict__ undo,
                                                 float* __restrict__ prob) {
  const int bh = blockIdx.y;
  const int t = blockIdx.x * 256 + threadIdx.x;
  const int* un = undo + (size_t)bh * (NHn * Tn);
  const float* ls = lse + (size_t)bh * (NHn * Tn);
  float* pr = prob + (size_t)bh * (NHn * Tn);
  int s[NHn];
  float v[NHn];
  float m = -3.0e38f;
#pragma unroll
  for (int h = 0; h < NHn; ++h) {
    s[h] = un[h * Tn + t];
    v[h] = ls[s[h]];
    m = fmaxf(m, v[h]);
  }
  float z = 0.f;
#pragma unroll
  for (int h = 0; h < NHn; ++h) z += expf(v[h] - m);
  float iz = 1.f / z;
#pragma unroll
  for (int h = 0; h < NHn; ++h) pr[s[h]] = expf(v[h] - m) * iz;
}

// ---------------------------------------------------------------------------
// Attention pass 2: recompute dots, P = exp(dots - lse) (bf16, LDS), O = P V (WMMA),
// scale by per-hash prob and atomically accumulate into [B,T,H*DH] f32.
// grid (512, 16), block 128
// ---------------------------------------------------------------------------
__global__ __launch_bounds__(128) void k_attn_out(const unsigned short* __restrict__ qk,
                                                  const unsigned short* __restrict__ vv,
                                                  const int* __restrict__ st,
                                                  const float* __restrict__ lse,
                                                  const float* __restrict__ prob,
                                                  float* __restrict__ ocomb) {
  __shared__ __align__(16) unsigned char smA[64 * 136 * 2];   // Q tile, then P tile
  __shared__ __align__(16) unsigned char smB[128 * 136 * 2];  // K tile, then V tile
  __shared__ int qpos[64];
  __shared__ int kpos[128];
  __shared__ float lse_sh[64];
  __shared__ float prob_sh[64];
  unsigned short* Qs = (unsigned short*)smA;
  unsigned short* Ks = (unsigned short*)smB;
  const int c = blockIdx.x, bh = blockIdx.y;
  const int pc = (c + NCn - 1) & (NCn - 1);
  const int tid = threadIdx.x, lane = tid & 31, w = tid >> 5;
  const int b_ = bh >> 3, hd = bh & 7;
  const int* stR = st + (size_t)bh * (NHn * Tn);
  if (tid < 64) {
    qpos[tid] = stR[c * 64 + tid];
    lse_sh[tid] = lse[(size_t)bh * (NHn * Tn) + c * 64 + tid];
    prob_sh[tid] = prob[(size_t)bh * (NHn * Tn) + c * 64 + tid];
  }
  kpos[tid] = (tid < 64) ? stR[c * 64 + tid] : stR[pc * 64 + (tid - 64)];
  __syncthreads();
  if (tid < 64) {
    const us8* src = (const us8*)(qk + ((size_t)(b_ * Tn + qpos[tid])) * (Hn * DHn) + hd * DHn);
    us8* dst = (us8*)&Qs[tid * 136];
#pragma unroll
    for (int d8 = 0; d8 < DHn / 8; ++d8) dst[d8] = src[d8];
  }
  {
    const us8* src = (const us8*)(qk + ((size_t)(b_ * Tn + kpos[tid])) * (Hn * DHn) + hd * DHn);
    us8* dst = (us8*)&Ks[tid * 136];
    float nrm = 0.f;
#pragma unroll
    for (int d8 = 0; d8 < DHn / 8; ++d8) {
      us8 vld = src[d8];
      dst[d8] = vld;
#pragma unroll
      for (int e = 0; e < 8; ++e) { float f = bf2f(vld[e]); nrm += f * f; }
    }
    float inv = 1.f / fmaxf(sqrtf(nrm), 1e-12f);
#pragma unroll
    for (int d8 = 0; d8 < DHn / 8; ++d8) {
      us8 vld = dst[d8];
#pragma unroll
      for (int e = 0; e < 8; ++e) vld[e] = f2bf(bf2f(vld[e]) * inv);
      dst[d8] = vld;
    }
  }
  __syncthreads();
  v8f acc[8];
  for (int nt = 0; nt < 8; ++nt)
    for (int e = 0; e < 8; ++e) acc[nt][e] = 0.f;
  for (int k0 = 0; k0 < DHn; k0 += 32) {
    v16bf a = fragA(Qs, 136, w * 16, k0, lane);
#pragma unroll
    for (int nt = 0; nt < 8; ++nt) {
      v16bf bfr = fragB_NK(Ks, 136, nt * 16, k0, lane);
      acc[nt] = wmma_bf16(a, bfr, acc[nt]);
    }
  }
  __syncthreads();  // done reading Q,K -> reuse buffers
  unsigned short* Ps = (unsigned short*)smA;
  unsigned short* Vs = (unsigned short*)smB;
  const int half8 = (lane & 16) ? 8 : 0;
  const float scale = 0.08838834764831845f;
#pragma unroll
  for (int nt = 0; nt < 8; ++nt)
#pragma unroll
    for (int r = 0; r < 8; ++r) {
      int i = w * 16 + r + half8;
      int j = nt * 16 + (lane & 15);
      float d = acc[nt][r] * scale;
      if (qpos[i] < kpos[j]) d = -1e9f;
      if (qpos[i] == kpos[j]) d = -5e4f;
      Ps[i * 136 + j] = f2bf(expf(d - lse_sh[i]));
    }
  {  // load V rows (key-index major = K x N for the PV matmul)
    const us8* src = (const us8*)(vv + ((size_t)(b_ * Tn + kpos[tid])) * (Hn * DHn) + hd * DHn);
    us8* dst = (us8*)&Vs[tid * 136];
#pragma unroll
    for (int d8 = 0; d8 < DHn / 8; ++d8) dst[d8] = src[d8];
  }
  __syncthreads();
  v8f oacc[8];
  for (int nt = 0; nt < 8; ++nt)
    for (int e = 0; e < 8; ++e) oacc[nt][e] = 0.f;
  for (int k0 = 0; k0 < 128; k0 += 32) {
    v16bf a = fragA(Ps, 136, w * 16, k0, lane);
#pragma unroll
    for (int nt = 0; nt < 8; ++nt) {
      v16bf bfr = fragB_KN(Vs, 136, k0, nt * 16, lane);
      oacc[nt] = wmma_bf16(a, bfr, oacc[nt]);
    }
  }
#pragma unroll
  for (int nt = 0; nt < 8; ++nt)
#pragma unroll
    for (int r = 0; r < 8; ++r) {
      int i = w * 16 + r + half8;
      int j = nt * 16 + (lane & 15);
      float val = oacc[nt][r] * prob_sh[i];
      int tpos = qpos[i];
      atomicAdd(&ocomb[((size_t)(b_ * Tn + tpos)) * (Hn * DHn) + hd * DHn + j], val);
    }
}

// ---------------------------------------------------------------------------
__global__ void k_f2b(const float* __restrict__ a, unsigned short* __restrict__ o, size_t n) {
  size_t i = ((size_t)blockIdx.x * 256 + threadIdx.x) * 4;
  if (i < n) {
    f4 v = *(const f4*)&a[i];
    us4 r;
#pragma unroll
    for (int e = 0; e < 4; ++e) r[e] = f2bf(v[e]);
    *(us4*)&o[i] = r;
  }
}

__global__ __launch_bounds__(256) void k_pool(const float* __restrict__ x1,
                                              const float* __restrict__ x2,
                                              float* __restrict__ out) {
  const int b = blockIdx.y;
  const int d = blockIdx.x * 256 + threadIdx.x;
  float s = 0.f;
  for (int t = 0; t < Tn; ++t) {
    size_t idx = ((size_t)(b * Tn + t)) * Dn + d;
    s += x1[idx] + x2[idx];
  }
  out[b * Dn + d] = s * (0.5f / Tn);
}

// ---------------------------------------------------------------------------
extern "C" void kernel_launch(void* const* d_in, const int* in_sizes, int n_in,
                              void* d_out, int out_size, void* d_ws, size_t ws_size,
                              hipStream_t stream) {
  (void)in_sizes; (void)n_in; (void)out_size; (void)ws_size;
  const float* x    = (const float*)d_in[0];
  const float* Wqk  = (const float*)d_in[1];
  const float* Wv   = (const float*)d_in[2];
  const float* Wout = (const float*)d_in[3];
  const float* bout = (const float*)d_in[4];
  const float* ln1g = (const float*)d_in[5];
  const float* ln1b = (const float*)d_in[6];
  const float* W1   = (const float*)d_in[7];
  const float* b1   = (const float*)d_in[8];
  const float* W2   = (const float*)d_in[9];
  const float* b2   = (const float*)d_in[10];
  const float* ln2g = (const float*)d_in[11];
  const float* ln2b = (const float*)d_in[12];
  const float* rots = (const float*)d_in[13];

  unsigned char* p = (unsigned char*)d_ws;
  auto take = [&](size_t bytes) -> void* {
    void* r = (void*)p;
    p += (bytes + 255) & ~(size_t)255;
    return r;
  };
  const size_t nTok = (size_t)Bn * Tn;  // 8192
  float* x1    = (float*)take(nTok * Dn * 4);
  float* x2    = (float*)take(nTok * Dn * 4);
  float* ocomb = (float*)take(nTok * Dn * 4);
  unsigned short* hbf  = (unsigned short*)take(nTok * Dn * 2);
  unsigned short* qkbf = (unsigned short*)take(nTok * Dn * 2);
  unsigned short* vbf  = (unsigned short*)take(nTok * Dn * 2);
  unsigned short* obf  = (unsigned short*)take(nTok * Dn * 2);
  unsigned short* gbf  = (unsigned short*)take(nTok * FFn * 2);
  int*   buckets = (int*)take((size_t)BHn * NHn * Tn * 4);
  int*   st      = (int*)take((size_t)BHn * NHn * Tn * 4);
  int*   undo    = (int*)take((size_t)BHn * NHn * Tn * 4);
  float* lse     = (float*)take((size_t)BHn * NHn * Tn * 4);
  float* prob    = (float*)take((size_t)BHn * NHn * Tn * 4);
  unsigned short* WqkB  = (unsigned short*)take((size_t)Ln * Dn * Dn * 2);
  unsigned short* WvB   = (unsigned short*)take((size_t)Ln * Dn * Dn * 2);
  unsigned short* WoutB = (unsigned short*)take((size_t)Ln * Dn * Dn * 2);
  unsigned short* W1B   = (unsigned short*)take((size_t)Ln * Dn * FFn * 2);
  unsigned short* W2B   = (unsigned short*)take((size_t)Ln * FFn * Dn * 2);

  // reversible streams x1 = x2 = x
  hipMemcpyAsync(x1, x, nTok * Dn * 4, hipMemcpyDeviceToDevice, stream);
  hipMemcpyAsync(x2, x, nTok * Dn * 4, hipMemcpyDeviceToDevice, stream);

  auto cvt = [&](const float* src, unsigned short* dst, size_t n) {
    k_f2b<<<dim3((unsigned)((n / 4 + 255) / 256)), 256, 0, stream>>>(src, dst, n);
  };
  cvt(Wqk,  WqkB,  (size_t)Ln * Dn * Dn);
  cvt(Wv,   WvB,   (size_t)Ln * Dn * Dn);
  cvt(Wout, WoutB, (size_t)Ln * Dn * Dn);
  cvt(W1,   W1B,   (size_t)Ln * Dn * FFn);
  cvt(W2,   W2B,   (size_t)Ln * FFn * Dn);

  for (int l = 0; l < Ln; ++l) {
    const size_t wD2 = (size_t)l * Dn * Dn;
    const size_t wDF = (size_t)l * Dn * FFn;
    // ---- attention block on x2, residual into x1 ----
    k_ln<<<dim3((unsigned)nTok), 256, 0, stream>>>(x2, ln1g + l * Dn, ln1b + l * Dn, hbf);
    k_gemm<2><<<dim3(Dn / 128, (unsigned)(nTok / 128)), 256, 0, stream>>>(
        hbf, WqkB + wD2, nullptr, nullptr, nullptr, qkbf, (int)nTok, Dn, Dn);
    k_gemm<2><<<dim3(Dn / 128, (unsigned)(nTok / 128)), 256, 0, stream>>>(
        hbf, WvB + wD2, nullptr, nullptr, nullptr, vbf, (int)nTok, Dn, Dn);
    k_buckets<<<dim3(Tn / 64, NHn, BHn), 64, 0, stream>>>(
        qkbf, rots + (size_t)l * DHn * NHn * Rn, buckets);
    k_sort<<<dim3(NHn, BHn), 256, 0, stream>>>(buckets, st, undo);
    k_attn_lse<<<dim3(NCn, BHn), 128, 0, stream>>>(qkbf, st, lse);
    k_combine<<<dim3(Tn / 256, BHn), 256, 0, stream>>>(lse, undo, prob);
    hipMemsetAsync(ocomb, 0, nTok * Dn * 4, stream);
    k_attn_out<<<dim3(NCn, BHn), 128, 0, stream>>>(qkbf, vbf, st, lse, prob, ocomb);
    k_f2b<<<dim3((unsigned)((nTok * Dn / 4 + 255) / 256)), 256, 0, stream>>>(ocomb, obf, nTok * Dn);
    k_gemm<1><<<dim3(Dn / 128, (unsigned)(nTok / 128)), 256, 0, stream>>>(
        obf, WoutB + wD2, bout + l * Dn, x1, x1, nullptr, (int)nTok, Dn, Dn);
    // ---- FF block on x1, residual into x2 ----
    k_ln<<<dim3((unsigned)nTok), 256, 0, stream>>>(x1, ln2g + l * Dn, ln2b + l * Dn, hbf);
    k_gemm<3><<<dim3(FFn / 128, (unsigned)(nTok / 128)), 256, 0, stream>>>(
        hbf, W1B + wDF, b1 + (size_t)l * FFn, nullptr, nullptr, gbf, (int)nTok, FFn, Dn);
    k_gemm<1><<<dim3(Dn / 128, (unsigned)(nTok / 128)), 256, 0, stream>>>(
        gbf, W2B + wDF, b2 + l * Dn, x2, x2, nullptr, (int)nTok, Dn, FFn);
  }
  k_pool<<<dim3(Dn / 256, Bn), 256, 0, stream>>>(x1, x2, (float*)d_out);
}
